// Predictor_82901458747654
// MI455X (gfx1250) — matrix operand order repaired and checked by reading, not compiled, not run on hardware
//
#include <hip/hip_runtime.h>
#include <stdint.h>

// ---------------------------------------------------------------------------
// MI455X (gfx1250) implementation — round 2.
// * All GEMM stages on v_wmma_f32_16x16x32_f16 (wave32, f32 accum).
// * Activations stored COLUMN-MAJOR ([col = b*128+l][channels], f16) and the
//   conv reduction ordered tap-major (k = tap*Cin + cin) so every B fragment
//   is a contiguous 32B run per lane -> global_load_b128, no scalar gathers.
// * MT-way register blocking: one B fragment feeds MT WMMAs.
// * lw stage (52 MB f16 stream) is split-K x16 for bandwidth, reduced by a
//   deterministic f32 reduction kernel (no atomics).
// Shapes: B=32, L=128, D=8, K=5, H=200, 3H=600 (pad 608), cols = B*L = 4096.
// ---------------------------------------------------------------------------

typedef __attribute__((ext_vector_type(16))) _Float16 v16h;
typedef __attribute__((ext_vector_type(8)))  float    v8f;

union Frag {
    v16h     h;
    uint32_t u[8];
};

__device__ __forceinline__ float lrelu01(float v) {
    return v > 0.0f ? v : 0.01f * v;
}

// A fragment (16x32 f16): lane m = lane&15; K pairs split by lane-half.
__device__ __forceinline__ void loadA(Frag& A, const uint16_t* row, int kb, int half) {
#pragma unroll
    for (int v = 0; v < 8; ++v) {
        int k = kb + half * 16 + 2 * v;
        A.u[v] = *reinterpret_cast<const uint32_t*>(row + k);
    }
}

// ---------------------------------------------------------------------------
// Weight pack: fp32 (Dcnt, Msrc, Cin, taps) -> f16 (Dcnt, Mpad, Kpad),
// reduction reordered TAP-MAJOR: dst k = tap*Cin + cin. Zero-padded.
// ---------------------------------------------------------------------------
__global__ void pack_w_tapmajor(const float* __restrict__ src, _Float16* __restrict__ dst,
                                int Dcnt, int Msrc, int Mpad, int Cin, int taps, int Kpad) {
    size_t total = (size_t)Dcnt * Mpad * Kpad;
    for (size_t i = (size_t)blockIdx.x * blockDim.x + threadIdx.x; i < total;
         i += (size_t)gridDim.x * blockDim.x) {
        int    k = (int)(i % Kpad);
        size_t r = i / Kpad;
        int    m = (int)(r % Mpad);
        int    d = (int)(r / Mpad);
        int    tap = k / Cin;
        int    cin = k - tap * Cin;
        float  v = 0.0f;
        if (m < Msrc && tap < taps)
            v = src[((size_t)d * Msrc + m) * ((size_t)Cin * taps) + (size_t)cin * taps + tap];
        dst[i] = (_Float16)v;
    }
}

// Row-major K pack (for w_hh as WMMA-B rows, and for lw): zero-padded.
__global__ void pack_pad_f16(const float* __restrict__ src, _Float16* __restrict__ dst,
                             int Dcnt, int Msrc, int Mpad, int Ksrc, int Kpad) {
    size_t total = (size_t)Dcnt * Mpad * Kpad;
    for (size_t i = (size_t)blockIdx.x * blockDim.x + threadIdx.x; i < total;
         i += (size_t)gridDim.x * blockDim.x) {
        int    k = (int)(i % Kpad);
        size_t r = i / Kpad;
        int    m = (int)(r % Mpad);
        int    d = (int)(r / Mpad);
        float  v = 0.0f;
        if (m < Msrc && k < Ksrc) v = src[((size_t)d * Msrc + m) * Ksrc + k];
        dst[i] = (_Float16)v;
    }
}

// x: (B, L, 2D) fp32 -> act0 col-major: [d][col][c(2)] f16
__global__ void pack_x_kernel(const float* __restrict__ x, _Float16* __restrict__ act0) {
    int i = blockIdx.x * blockDim.x + threadIdx.x;   // < 8*4096*2
    if (i >= 8 * 4096 * 2) return;
    int c   = i & 1;
    int col = (i >> 1) & 4095;
    int d   = i >> 13;
    int b   = col >> 7, l = col & 127;
    act0[i] = (_Float16)x[((size_t)b * 128 + l) * 16 + 2 * d + c];
}

// ---------------------------------------------------------------------------
// Tap-uniform implicit-GEMM (requires Cin % 32 == 0). One wave computes MT
// 16x16 output tiles sharing one B fragment per K-chunk.
//   act : [D][4096][Cin]  col-major f16
//   wp  : [D][Mpad][TAPS*Cin] tap-major f16
//   out : [D][4096][Mpad] col-major (f16 and/or f32)
// ---------------------------------------------------------------------------
template <int TAPS, int MT>
__global__ __launch_bounds__(32)
void wmma_gemm_cm(const _Float16* __restrict__ act, const _Float16* __restrict__ wp,
                  const float* __restrict__ bias, _Float16* __restrict__ outH,
                  float* __restrict__ outF, int Cin, int Mrows, int Mpad, int applyLrelu) {
    const int lane = threadIdx.x;
    const int ntile = blockIdx.x;              // 0..255
    const int mg    = blockIdx.y;              // m-group of MT tiles
    const int d     = blockIdx.z;
    const int nloc  = lane & 15;
    const int half  = lane >> 4;
    const int col   = ntile * 16 + nloc;
    const int bi    = col >> 7, li = col & 127;
    const int pad   = (TAPS - 1) >> 1;
    const int Kpad  = TAPS * Cin;

    const uint16_t* actd = (const uint16_t*)act + (size_t)d * 4096 * Cin;
    const uint16_t* wrow[MT];
#pragma unroll
    for (int mt = 0; mt < MT; ++mt)
        wrow[mt] = (const uint16_t*)wp +
                   ((size_t)d * Mpad + (size_t)(mg * MT + mt) * 16 + nloc) * Kpad;

    v8f acc[MT];
#pragma unroll
    for (int mt = 0; mt < MT; ++mt) acc[mt] = (v8f){};

    const int nk = Kpad >> 5;
    for (int kc = 0; kc < nk; ++kc) {
        const int kb   = kc << 5;
        const int tap  = kb / Cin;             // uniform within chunk (Cin%32==0)
        const int cinb = kb - tap * Cin;
        const int sl   = li + tap - pad;
        const bool ok  = (sl >= 0) && (sl < 128);
        const uint16_t* base = actd + ((size_t)(bi * 128 + sl)) * Cin + cinb + half * 16;

        Frag Bf;
#pragma unroll
        for (int v = 0; v < 8; ++v)
            Bf.u[v] = ok ? *reinterpret_cast<const uint32_t*>(base + 2 * v) : 0u;

#pragma unroll
        for (int mt = 0; mt < MT; ++mt) {
            Frag A;
            loadA(A, wrow[mt], kb, half);
            acc[mt] = __builtin_amdgcn_wmma_f32_16x16x32_f16(false, A.h, false, Bf.h,
                                                             (short)0, acc[mt], false, false);
        }
    }

    const size_t obase = ((size_t)d * 4096 + col) * Mpad;
#pragma unroll
    for (int mt = 0; mt < MT; ++mt) {
#pragma unroll
        for (int v = 0; v < 8; ++v) {
            int   m   = (mg * MT + mt) * 16 + v + 8 * half;
            float val = acc[mt][v];
            if (m < Mrows) val += bias[(size_t)d * Mrows + m];
            if (applyLrelu) val = lrelu01(val);
            if (outH) outH[obase + m] = (_Float16)val;   // 8 contiguous f16 -> b128
            if (outF) outF[obase + m] = val;
        }
    }
}

// conv1 special case: Cin=2, taps=5, K padded to 32 (one chunk), Mpad=64.
__global__ __launch_bounds__(32)
void wmma_conv1_cm(const _Float16* __restrict__ act0,  // [D][4096][2]
                   const _Float16* __restrict__ wp,    // [D][64][32] tap-major
                   const float* __restrict__ bias,     // [D][64]
                   _Float16* __restrict__ out) {       // [D][4096][64]
    const int lane = threadIdx.x;
    const int ntile = blockIdx.x;
    const int d     = blockIdx.z;
    const int nloc  = lane & 15;
    const int half  = lane >> 4;
    const int col   = ntile * 16 + nloc;
    const int bi    = col >> 7, li = col & 127;

    const uint16_t* actd = (const uint16_t*)act0 + (size_t)d * 4096 * 2;
    Frag Bf;
#pragma unroll
    for (int v = 0; v < 8; ++v) {
        uint32_t pair = 0;
#pragma unroll
        for (int s = 0; s < 2; ++s) {
            int k = half * 16 + 2 * v + s;     // tap-major: tap = k>>1, cin = k&1
            int tap = k >> 1, cin = k & 1;
            int sl = li + tap - 2;
            uint16_t bits = 0;
            if (k < 10 && sl >= 0 && sl < 128)
                bits = actd[((size_t)(bi * 128 + sl)) * 2 + cin];
            pair |= ((uint32_t)bits) << (16 * s);
        }
        Bf.u[v] = pair;
    }

    v8f acc[4];
#pragma unroll
    for (int mt = 0; mt < 4; ++mt) acc[mt] = (v8f){};
#pragma unroll
    for (int mt = 0; mt < 4; ++mt) {
        Frag A;
        loadA(A, (const uint16_t*)wp + ((size_t)d * 64 + (size_t)mt * 16 + nloc) * 32, 0, half);
        acc[mt] = __builtin_amdgcn_wmma_f32_16x16x32_f16(false, A.h, false, Bf.h,
                                                         (short)0, acc[mt], false, false);
    }

    const size_t obase = ((size_t)d * 4096 + col) * 64;
#pragma unroll
    for (int mt = 0; mt < 4; ++mt)
#pragma unroll
        for (int v = 0; v < 8; ++v) {
            int m = mt * 16 + v + 8 * half;
            out[obase + m] = (_Float16)lrelu01(acc[mt][v] + bias[(size_t)d * 64 + m]);
        }
}

// ---------------------------------------------------------------------------
// lw projection, split-K x16 (bandwidth stage: streams 52 MB of f16 weights).
// partial[sp][d][t(128)][b(32)] f32.
// ---------------------------------------------------------------------------
__global__ __launch_bounds__(32)
void wmma_gemm_lw_partial(const _Float16* __restrict__ fbuf,  // [D][32][25600]
                          const _Float16* __restrict__ lwp,   // [D][128][25600]
                          float* __restrict__ partial) {
    const int lane  = threadIdx.x;
    const int ntile = blockIdx.x;          // 0..1
    const int mtile = blockIdx.y;          // 0..7
    const int d     = blockIdx.z >> 4;
    const int sp    = blockIdx.z & 15;
    const int nloc  = lane & 15;
    const int half  = lane >> 4;

    const uint16_t* wrow = (const uint16_t*)lwp +
                           ((size_t)d * 128 + (size_t)mtile * 16 + nloc) * 25600;
    const uint16_t* frow = (const uint16_t*)fbuf +
                           ((size_t)d * 32 + (size_t)ntile * 16 + nloc) * 25600;

    v8f acc = {};
    const int kc0 = sp * 50;
    for (int kc = kc0; kc < kc0 + 50; ++kc) {
        const int kb = kc << 5;
        __builtin_prefetch(wrow + kb + 256, 0, 1);   // global_prefetch_b8
        __builtin_prefetch(frow + kb + 256, 0, 1);
        Frag A, Bf;
        loadA(A, wrow, kb, half);
#pragma unroll
        for (int v = 0; v < 8; ++v)
            Bf.u[v] = *reinterpret_cast<const uint32_t*>(frow + kb + half * 16 + 2 * v);
        acc = __builtin_amdgcn_wmma_f32_16x16x32_f16(false, A.h, false, Bf.h,
                                                     (short)0, acc, false, false);
    }

#pragma unroll
    for (int v = 0; v < 8; ++v) {
        int m = mtile * 16 + v + 8 * half;      // t
        int n = ntile * 16 + nloc;              // batch
        partial[(((size_t)sp * 8 + d) * 128 + m) * 32 + n] = acc[v];
    }
}

// Reduce split-K partials -> seqB col-major [4096][32] f16 (channels 8..31 = 0).
__global__ void seq_reduce(const float* __restrict__ partial, const float* __restrict__ lb,
                           _Float16* __restrict__ seqB) {
    int i = blockIdx.x * blockDim.x + threadIdx.x;   // < 4096*32
    if (i >= 4096 * 32) return;
    int ch  = i & 31;
    int col = i >> 5;
    float val = 0.0f;
    if (ch < 8) {
        int d = ch, b = col >> 7, t = col & 127;
        float s = 0.0f;
        for (int sp = 0; sp < 16; ++sp)
            s += partial[(((size_t)sp * 8 + d) * 128 + t) * 32 + b];
        val = lrelu01(s + lb[(size_t)d * 128 + t]);
    }
    seqB[i] = (_Float16)val;
}

// ---------------------------------------------------------------------------
// GRU recurrence (gates r,z,n). One 16-wave workgroup per dim.
// LDS: h (32x224 f16) + gh (32x608 f32) = 92160 B. xw is COL-MAJOR [4096][608].
// ---------------------------------------------------------------------------
__global__ __launch_bounds__(512)
void gru_recurrent(const float*    __restrict__ xw,     // [D][4096][608] (incl b_ih)
                   const _Float16* __restrict__ whh,    // [D][608][224]
                   const float*    __restrict__ bhh,    // [D][600]
                   _Float16*       __restrict__ fOut,   // [D][32][25600] (lrelu) or null
                   float*          __restrict__ flatOut) { // [32][25600] (lrelu) or null
    extern __shared__ unsigned char smem[];
    _Float16* hS = (_Float16*)smem;                  // 32 x 224
    float*    gh = (float*)(smem + 32 * 224 * 2);    // 32 x 608

    const int d    = blockIdx.x;
    const int tid  = threadIdx.x;
    const int wave = tid >> 5;
    const int lane = tid & 31;
    const int nloc = lane & 15;
    const int half = lane >> 4;

    for (int i = tid; i < 32 * 224; i += 512) hS[i] = (_Float16)0.0f;
    __syncthreads();

    const uint16_t* whd = (const uint16_t*)whh + (size_t)d * 608 * 224;
    const float*    xwd = xw + (size_t)d * 4096 * 608;
    const float*    bhd = bhh + (size_t)d * 600;

    for (int t = 0; t < 128; ++t) {
        for (int tile = wave; tile < 76; tile += 16) {     // wave-uniform
            int mt = tile & 1;
            int nt = tile >> 1;
            const uint16_t* hrow = (const uint16_t*)hS + ((size_t)mt * 16 + nloc) * 224;
            const uint16_t* wrow = whd + ((size_t)nt * 16 + nloc) * 224;
            v8f acc = {};
#pragma unroll
            for (int kc = 0; kc < 7; ++kc) {
                const int kb = kc << 5;
                Frag A, Bf;
                loadA(A, hrow, kb, half);
#pragma unroll
                for (int v = 0; v < 8; ++v)
                    Bf.u[v] = *reinterpret_cast<const uint32_t*>(wrow + kb + half * 16 + 2 * v);
                acc = __builtin_amdgcn_wmma_f32_16x16x32_f16(false, A.h, false, Bf.h,
                                                             (short)0, acc, false, false);
            }
#pragma unroll
            for (int v = 0; v < 8; ++v)
                gh[((size_t)mt * 16 + v + 8 * half) * 608 + nt * 16 + nloc] = acc[v];
        }
        __syncthreads();

        for (int i = tid; i < 32 * 200; i += 512) {
            int b = i / 200, j = i - b * 200;
            int col = (b << 7) + t;
            const float* xcol = xwd + (size_t)col * 608;
            float xr = xcol[j];
            float xz = xcol[200 + j];
            float xn = xcol[400 + j];
            float hr = gh[b * 608 + j]       + bhd[j];
            float hz = gh[b * 608 + 200 + j] + bhd[200 + j];
            float hn = gh[b * 608 + 400 + j] + bhd[400 + j];
            float r  = 1.0f / (1.0f + __expf(-(xr + hr)));
            float z  = 1.0f / (1.0f + __expf(-(xz + hz)));
            float nn = tanhf(xn + r * hn);
            float hp = (float)hS[b * 224 + j];
            float hv = (1.0f - z) * nn + z * hp;
            hS[b * 224 + j] = (_Float16)hv;
            float lr = lrelu01(hv);
            if (fOut)    fOut[((size_t)d * 32 + b) * 25600 + (size_t)t * 200 + j] = (_Float16)lr;
            if (flatOut) flatOut[(size_t)b * 25600 + (size_t)t * 200 + j] = lr;
        }
        __syncthreads();
    }
}

// Final linear: out[b][dm] = lrelu(dot(flat[b], last_w[dm]) + last_b[dm]).
__global__ __launch_bounds__(256)
void last_kernel(const float* __restrict__ flat, const float* __restrict__ lastw,
                 const float* __restrict__ lastb, float* __restrict__ out) {
    __shared__ float red[256];
    int b  = blockIdx.x >> 3;
    int dm = blockIdx.x & 7;
    float s = 0.0f;
    for (int k = threadIdx.x; k < 25600; k += 256)
        s += flat[(size_t)b * 25600 + k] * lastw[(size_t)dm * 25600 + k];
    red[threadIdx.x] = s;
    __syncthreads();
    for (int st = 128; st > 0; st >>= 1) {
        if (threadIdx.x < st) red[threadIdx.x] += red[threadIdx.x + st];
        __syncthreads();
    }
    if (threadIdx.x == 0) out[b * 8 + dm] = lrelu01(red[0] + lastb[dm]);
}

// ---------------------------------------------------------------------------
extern "C" void kernel_launch(void* const* d_in, const int* in_sizes, int n_in,
                              void* d_out, int out_size, void* d_ws, size_t ws_size,
                              hipStream_t stream) {
    (void)in_sizes; (void)n_in; (void)out_size; (void)ws_size;

    const float* x      = (const float*)d_in[0];
    const float* c1w    = (const float*)d_in[1];
    const float* c1b    = (const float*)d_in[2];
    const float* c2w    = (const float*)d_in[3];
    const float* c2b    = (const float*)d_in[4];
    const float* c3w    = (const float*)d_in[5];
    const float* c3b    = (const float*)d_in[6];
    const float* c4w    = (const float*)d_in[7];
    const float* c4b    = (const float*)d_in[8];
    const float* gwi    = (const float*)d_in[9];
    const float* gwh    = (const float*)d_in[10];
    const float* gbi    = (const float*)d_in[11];
    const float* gbh    = (const float*)d_in[12];
    const float* lw     = (const float*)d_in[13];
    const float* lb     = (const float*)d_in[14];
    const float* rwi    = (const float*)d_in[15];
    const float* rwh    = (const float*)d_in[16];
    const float* rbi    = (const float*)d_in[17];
    const float* rbh    = (const float*)d_in[18];
    const float* last_w = (const float*)d_in[19];
    const float* last_b = (const float*)d_in[20];
    float* out = (float*)d_out;

    char*  ws  = (char*)d_ws;
    size_t off = 0;
    auto alloc = [&](size_t bytes) -> size_t {
        size_t o = off;
        off = (off + bytes + 255) & ~(size_t)255;
        return o;
    };
    size_t o_act0 = alloc((size_t)8 * 4096 * 2   * 2);
    size_t o_act1 = alloc((size_t)8 * 4096 * 64  * 2);
    size_t o_act2 = alloc((size_t)8 * 4096 * 128 * 2);
    size_t o_act3 = alloc((size_t)8 * 4096 * 256 * 2);
    size_t o_act4 = alloc((size_t)8 * 4096 * 512 * 2);
    size_t o_w1p  = alloc((size_t)8 * 64  * 32   * 2);
    size_t o_w2p  = alloc((size_t)8 * 128 * 320  * 2);
    size_t o_w3p  = alloc((size_t)8 * 256 * 640  * 2);
    size_t o_w4p  = alloc((size_t)8 * 512 * 1280 * 2);
    size_t o_gwip = alloc((size_t)8 * 608 * 512  * 2);
    size_t o_whhp = alloc((size_t)8 * 608 * 224  * 2);
    size_t o_xw   = alloc((size_t)8 * 4096 * 608 * 4);   // f32; reused for lwPack later
    size_t o_fbuf = alloc((size_t)8 * 32 * 25600 * 2);
    size_t o_part = alloc((size_t)16 * 8 * 128 * 32 * 4);
    size_t o_seqB = alloc((size_t)4096 * 32 * 2);
    size_t o_rwip = alloc((size_t)608 * 32  * 2);
    size_t o_rwhp = alloc((size_t)608 * 224 * 2);
    size_t o_xw2  = alloc((size_t)4096 * 608 * 4);
    size_t o_flat = alloc((size_t)32 * 25600 * 4);
    size_t o_lwp  = o_xw;   // alias: xw dead after per-dim recurrences

    _Float16* act0 = (_Float16*)(ws + o_act0);
    _Float16* act1 = (_Float16*)(ws + o_act1);
    _Float16* act2 = (_Float16*)(ws + o_act2);
    _Float16* act3 = (_Float16*)(ws + o_act3);
    _Float16* act4 = (_Float16*)(ws + o_act4);
    _Float16* w1p  = (_Float16*)(ws + o_w1p);
    _Float16* w2p  = (_Float16*)(ws + o_w2p);
    _Float16* w3p  = (_Float16*)(ws + o_w3p);
    _Float16* w4p  = (_Float16*)(ws + o_w4p);
    _Float16* gwip = (_Float16*)(ws + o_gwip);
    _Float16* whhp = (_Float16*)(ws + o_whhp);
    float*    xwB  = (float*)   (ws + o_xw);
    _Float16* fbuf = (_Float16*)(ws + o_fbuf);
    float*    part = (float*)   (ws + o_part);
    _Float16* seqB = (_Float16*)(ws + o_seqB);
    _Float16* rwip = (_Float16*)(ws + o_rwip);
    _Float16* rwhp = (_Float16*)(ws + o_rwhp);
    float*    xw2  = (float*)   (ws + o_xw2);
    float*    flat = (float*)   (ws + o_flat);
    _Float16* lwp  = (_Float16*)(ws + o_lwp);

    auto packGrid = [](size_t total) { return dim3((unsigned)((total + 255) / 256)); };

    // ---- packs ----
    pack_x_kernel<<<packGrid(65536), 256, 0, stream>>>(x, act0);
    pack_w_tapmajor<<<packGrid((size_t)8*64*32),    256, 0, stream>>>(c1w, w1p,  8, 64,  64,  2,   5, 32);
    pack_w_tapmajor<<<packGrid((size_t)8*128*320),  256, 0, stream>>>(c2w, w2p,  8, 128, 128, 64,  5, 320);
    pack_w_tapmajor<<<packGrid((size_t)8*256*640),  256, 0, stream>>>(c3w, w3p,  8, 256, 256, 128, 5, 640);
    pack_w_tapmajor<<<packGrid((size_t)8*512*1280), 256, 0, stream>>>(c4w, w4p,  8, 512, 512, 256, 5, 1280);
    pack_w_tapmajor<<<packGrid((size_t)8*608*512),  256, 0, stream>>>(gwi, gwip, 8, 600, 608, 512, 1, 512);
    pack_w_tapmajor<<<packGrid((size_t)608*32),     256, 0, stream>>>(rwi, rwip, 1, 600, 608, 8,   1, 32);
    pack_pad_f16<<<packGrid((size_t)8*608*224),     256, 0, stream>>>(gwh, whhp, 8, 600, 608, 200, 224);
    pack_pad_f16<<<packGrid((size_t)608*224),       256, 0, stream>>>(rwh, rwhp, 1, 600, 608, 200, 224);

    // ---- conv stack (col-major, tap-uniform WMMA) ----
    wmma_conv1_cm<<<dim3(256, 1, 8), 32, 0, stream>>>(act0, w1p, c1b, act1);
    wmma_gemm_cm<5, 4><<<dim3(256, 2, 8), 32, 0, stream>>>(act1, w2p, c2b, act2, nullptr, 64,  128, 128, 1);
    wmma_gemm_cm<5, 4><<<dim3(256, 4, 8), 32, 0, stream>>>(act2, w3p, c3b, act3, nullptr, 128, 256, 256, 1);
    wmma_gemm_cm<5, 4><<<dim3(256, 8, 8), 32, 0, stream>>>(act3, w4p, c4b, act4, nullptr, 256, 512, 512, 1);

    // ---- per-dim GRU input projection: xw[col][608] f32 = act4 @ gwi^T + gbi ----
    wmma_gemm_cm<1, 2><<<dim3(256, 19, 8), 32, 0, stream>>>(act4, gwip, gbi, nullptr, xwB, 512, 600, 608, 0);

    // ---- per-dim GRU recurrence -> f = lrelu(g) f16 ----
    gru_recurrent<<<dim3(8), 512, 32*224*2 + 32*608*4, stream>>>(xwB, whhp, gbh, fbuf, nullptr);

    // ---- lw stage: pack (aliases dead xw), split-K GEMM, reduce -> seqB ----
    pack_pad_f16<<<packGrid((size_t)8*128*25600), 256, 0, stream>>>(lw, lwp, 8, 128, 128, 25600, 25600);
    wmma_gemm_lw_partial<<<dim3(2, 8, 128), 32, 0, stream>>>(fbuf, lwp, part);
    seq_reduce<<<packGrid((size_t)4096*32), 256, 0, stream>>>(part, lb, seqB);

    // ---- combining GRU ----
    wmma_gemm_cm<1, 2><<<dim3(256, 19, 1), 32, 0, stream>>>(seqB, rwip, rbi, nullptr, xw2, 32, 600, 608, 0);
    gru_recurrent<<<dim3(1), 512, 32*224*2 + 32*608*4, stream>>>(xw2, rwhp, rbh, nullptr, flat);

    // ---- final linear ----
    last_kernel<<<dim3(256), 256, 0, stream>>>(flat, last_w, last_b, out);
}
